// MultiHeadSelfAttention_8993661518197
// MI455X (gfx1250) — matrix operand order repaired and checked by reading, not compiled, not run on hardware
//
#include <hip/hip_runtime.h>
#include <hip/hip_bf16.h>

// ---------------------------------------------------------------------------
// MI455X (gfx1250) multi-head self-attention forward.
// bf16 WMMA everywhere; attention core fused into one kernel using the
// 320KB-per-WGP LDS so the [H,B,T,T] att output is written exactly once,
// with float4-vectorized softmax/store phase.
// ---------------------------------------------------------------------------

typedef __bf16 bf16_t;
typedef __attribute__((ext_vector_type(16))) __bf16 v16bf;
typedef __attribute__((ext_vector_type(8)))  __bf16 v8bf;
typedef __attribute__((ext_vector_type(4)))  __bf16 v4bf;
typedef __attribute__((ext_vector_type(8)))  float  v8f;
typedef __attribute__((ext_vector_type(4)))  float  v4f;

namespace {
constexpr int kB = 4, kT = 2048, kC = 1024, kH = 16, kD = 64;
constexpr int kM  = kB * kT;      // 8192 rows of x
constexpr int kN3 = 3 * kC;       // 3072 fused QKV columns
constexpr size_t kYElems = (size_t)kM * kC;

// workspace layout (bytes)
constexpr size_t OFF_XB = 0;                               // x as bf16 [M][C]
constexpr size_t SZ_XB  = (size_t)kM * kC * 2;
constexpr size_t OFF_WA = OFF_XB + SZ_XB;                  // W_attn^T bf16 [3C][C]
constexpr size_t SZ_WA  = (size_t)kN3 * kC * 2;
constexpr size_t OFF_WP = OFF_WA + SZ_WA;                  // W_proj^T bf16 [C][C]
constexpr size_t SZ_WP  = (size_t)kC * kC * 2;
constexpr size_t OFF_Q  = OFF_WP + SZ_WP;                  // q bf16 [B][H][T][D] (pre-scaled)
constexpr size_t SZ_HD  = (size_t)kB * kH * kT * kD * 2;
constexpr size_t OFF_K  = OFF_Q + SZ_HD;                   // k bf16 [B][H][T][D]
constexpr size_t OFF_VT = OFF_K + SZ_HD;                   // v^T bf16 [B][H][D][T]
constexpr size_t OFF_YP = OFF_VT + SZ_HD;                  // y_pre bf16 [B][T][C]
} // namespace

// ---- fragment helpers (contiguous-k layouts, wave32) -------------------------

// A (16x32 bf16, row-major source): lane l<16 -> row l, K = k0..k0+15
//                                   lane l>=16 -> row l-16, K = k0+16..k0+31.
__device__ __forceinline__ v16bf load_a_bf16(const bf16_t* __restrict__ A, int lda, int lane) {
  const int row = lane & 15, half = lane >> 4;
  const bf16_t* p = A + (size_t)row * lda + half * 16;
  v8bf lo = *(const v8bf*)(p);
  v8bf hi = *(const v8bf*)(p + 8);
  v16bf f;
#pragma unroll
  for (int i = 0; i < 8; ++i) { f[i] = lo[i]; f[i + 8] = hi[i]; }
  return f;
}

// B (32x16 bf16) loaded from B^T [N][K] row-major: lane l<16 -> col l, K=k0..k0+15
//                                                  lane l>=16 -> col l-16, K=k0+16..k0+31
__device__ __forceinline__ v16bf load_b_bf16(const bf16_t* __restrict__ Bt, int ldb, int lane) {
  const int col = lane & 15, half = lane >> 4;
  return *(const v16bf*)(Bt + (size_t)col * ldb + half * 16);
}

__device__ __forceinline__ v8f wmma_bf16(v16bf a, v16bf b, v8f c) {
  // D = A x B + C, f32 accumulate
  return __builtin_amdgcn_wmma_f32_16x16x32_bf16(false, a, false, b, (short)0, c, false, false);
}

__device__ __forceinline__ v8f vzero8() {
  v8f z;
#pragma unroll
  for (int i = 0; i < 8; ++i) z[i] = 0.0f;
  return z;
}

// ---- prep kernels ----------------------------------------------------------

__global__ void __launch_bounds__(256) cvt_f32_to_bf16_kernel(
    const float* __restrict__ src, bf16_t* __restrict__ dst, int n) {
  int i = blockIdx.x * 256 + threadIdx.x;
  if (i < n) dst[i] = (bf16_t)src[i];
}

// src [rows][cols] f32 -> dst [cols][rows] bf16
__global__ void __launch_bounds__(256) transpose_f32_to_bf16_kernel(
    const float* __restrict__ src, bf16_t* __restrict__ dst, int rows, int cols) {
  int i = blockIdx.x * 256 + threadIdx.x;
  if (i < rows * cols) {
    int r = i / cols, c = i - r * cols;
    dst[(size_t)c * rows + r] = (bf16_t)src[i];
  }
}

// ---- stage 1: fused QKV projection ----------------------------------------
// One wave -> 16x64 tile of [M=8192, N=3072]. Scatters into q/k/v^T head layout.
__global__ void __launch_bounds__(256) qkv_gemm_kernel(
    const bf16_t* __restrict__ xb, const bf16_t* __restrict__ WaT,
    const float* __restrict__ b_attn,
    bf16_t* __restrict__ qb, bf16_t* __restrict__ kb, bf16_t* __restrict__ vT) {
  const int lane = threadIdx.x & 31;
  const int widx = blockIdx.x * 8 + (threadIdx.x >> 5);
  const int tn = widx % 48;           // 3072 / 64
  const int tm = widx / 48;           // 8192 / 16
  const int m0 = tm * 16;
  const int n0 = tn * 64;
  const bf16_t* A = xb + (size_t)m0 * kC;

  v8f acc[4];
#pragma unroll
  for (int j = 0; j < 4; ++j) acc[j] = vzero8();

  for (int k0 = 0; k0 < kC; k0 += 32) {
    __builtin_prefetch(A + k0 + 256, 0, 1);   // global_prefetch_b8 on next A chunk
    v16bf a = load_a_bf16(A + k0, kC, lane);
#pragma unroll
    for (int j = 0; j < 4; ++j) {
      v16bf b = load_b_bf16(WaT + (size_t)(n0 + j * 16) * kC + k0, kC, lane);
      acc[j] = wmma_bf16(a, b, acc[j]);
    }
  }

  const int col = lane & 15, half = lane >> 4;
#pragma unroll
  for (int j = 0; j < 4; ++j) {
    const int n = n0 + j * 16 + col;
    const float bias = b_attn[n];
    const int sel = n >> 10;     // 0=q 1=k 2=v
    const int c = n & 1023;
    const int h = c >> 6;
    const int d = c & 63;
#pragma unroll
    for (int r = 0; r < 8; ++r) {
      const int m = m0 + r + half * 8;
      const int b4 = m >> 11;          // batch
      const int t = m & 2047;          // time
      const float v = acc[j][r] + bias;
      const size_t hd = ((size_t)(b4 * kH + h) * kT + t) * kD + d;
      if (sel == 0)      qb[hd] = (bf16_t)(v * 0.125f);   // fold 1/sqrt(D)
      else if (sel == 1) kb[hd] = (bf16_t)v;
      else vT[((size_t)(b4 * kH + h) * kD + d) * kT + t] = (bf16_t)v;  // transposed v
    }
  }
}

// ---- stage 2: fused scores + softmax + att@V ------------------------------
// One workgroup (8 waves) owns 16 query rows of one (h,b):
//   A) 16x2048 masked/ALiBi logit strip via WMMA -> LDS (f32, 128KB)
//   B) per-wave softmax (2 rows each, shfl_xor reductions, float4-vectorized);
//      att written to HBM exactly once with b128 stores; bf16 probs -> LDS
//   C) att@V with A-fragments from LDS, K split across waves (causal bound),
//      partials reduced through the dead logit strip.
__global__ void __launch_bounds__(256) fused_attn_kernel(
    const bf16_t* __restrict__ qb, const bf16_t* __restrict__ kb,
    const bf16_t* __restrict__ vT, const unsigned char* __restrict__ pad,
    float* __restrict__ att, bf16_t* __restrict__ ypre) {
  __shared__ float  s_logits[16 * 2048];   // 128 KB (aliased as AV partials later)
  __shared__ bf16_t s_probs [16 * 2048];   //  64 KB

  const int lane = threadIdx.x & 31;
  const int wave = threadIdx.x >> 5;
  const int it = blockIdx.x & 127;
  const int bh = blockIdx.x >> 7;          // bh = h*4 + b  (att layout [H,B,T,T])
  const int h = bh >> 2;
  const int b = bh & 3;
  const int i0 = it * 16;
  const int col = lane & 15, half = lane >> 4;
  const float mh = exp2f(-0.5f * (float)(h + 1));   // ALiBi slope (2^8)^(-(h+1)/16)

  // ---------------- Phase A: logits into LDS ----------------
  const bf16_t* qbase = qb + (size_t)(b * kH + h) * kT * kD;
  const bf16_t* kbase = kb + (size_t)(b * kH + h) * kT * kD;
  const v16bf a0 = load_a_bf16(qbase + (size_t)i0 * kD, kD, lane);
  const v16bf a1 = load_a_bf16(qbase + (size_t)i0 * kD + 32, kD, lane);

  for (int jt = wave; jt < 128; jt += 8) {
    const int j0 = jt * 16;
    const bool dead = (j0 > i0 + 15);      // wave-uniform: tile above diagonal
    v8f acc = vzero8();
    if (!dead) {
      v16bf b0 = load_b_bf16(kbase + (size_t)j0 * kD, kD, lane);
      v16bf b1 = load_b_bf16(kbase + (size_t)j0 * kD + 32, kD, lane);
      acc = wmma_bf16(a0, b0, acc);
      acc = wmma_bf16(a1, b1, acc);
    }
    const int s = j0 + col;
    const bool padm = pad[b * kT + s] != 0;
#pragma unroll
    for (int r = 0; r < 8; ++r) {
      const int row = r + half * 8;
      const int t = i0 + row;
      float v = (!dead && s <= t && !padm) ? acc[r] : -1e9f;
      v += mh * (float)(s - t);            // ALiBi added after masking (matches ref)
      s_logits[row * 2048 + s] = v;
    }
  }
  __syncthreads();

  // ---------------- Phase B: softmax, write att once (vectorized) ----------
#pragma unroll
  for (int rr = 0; rr < 2; ++rr) {
    const int row = wave * 2 + rr;
    const int t = i0 + row;

    // one ds_load_b128 pass; row chunks live in registers (16 x v4f / lane)
    const v4f* lrow = (const v4f*)(s_logits + row * 2048);   // 512 v4f per row
    v4f vals[16];
    float mx = -3.4e38f;
#pragma unroll
    for (int i = 0; i < 16; ++i) {
      vals[i] = lrow[lane + i * 32];
      mx = fmaxf(mx, fmaxf(fmaxf(vals[i][0], vals[i][1]),
                           fmaxf(vals[i][2], vals[i][3])));
    }
#pragma unroll
    for (int off = 16; off > 0; off >>= 1) mx = fmaxf(mx, __shfl_xor(mx, off, 32));

    float sum = 0.0f;
#pragma unroll
    for (int i = 0; i < 16; ++i) {
      v4f e;
      e[0] = __expf(vals[i][0] - mx);
      e[1] = __expf(vals[i][1] - mx);
      e[2] = __expf(vals[i][2] - mx);
      e[3] = __expf(vals[i][3] - mx);
      vals[i] = e;
      sum += (e[0] + e[1]) + (e[2] + e[3]);
    }
#pragma unroll
    for (int off = 16; off > 0; off >>= 1) sum += __shfl_xor(sum, off, 32);
    const float inv = 1.0f / sum;

    v4f*  grow = (v4f*)(att + ((size_t)bh * kT + t) * kT);   // b128 stores
    v4bf* prow = (v4bf*)(s_probs + row * 2048);              // ds_store_b64
#pragma unroll
    for (int i = 0; i < 16; ++i) {
      const v4f p = vals[i] * inv;
      grow[lane + i * 32] = p;             // the ONLY att HBM traffic
      v4bf pb;
      pb[0] = (bf16_t)p[0]; pb[1] = (bf16_t)p[1];
      pb[2] = (bf16_t)p[2]; pb[3] = (bf16_t)p[3];
      prow[lane + i * 32] = pb;
    }
  }
  __syncthreads();

  // ---------------- Phase C: y_pre = probs @ v ----------------
  const bf16_t* Bt = vT + (size_t)(b * kH + h) * kD * kT;   // [D][T]
  v8f acc[4];
#pragma unroll
  for (int j = 0; j < 4; ++j) acc[j] = vzero8();

  const int kbeg = wave * 256;
  const int klim = i0 + 16;                // causal: probs==0 for s>t
  const int kend = (kbeg + 256 < klim) ? kbeg + 256 : klim;
  for (int k0 = kbeg; k0 < kend; k0 += 32) {
    // A fragment straight from LDS probs (ds_load_b128)
    const bf16_t* p = s_probs + (lane & 15) * 2048 + k0 + half * 16;
    v8bf lo = *(const v8bf*)(p);
    v8bf hi = *(const v8bf*)(p + 8);
    v16bf a;
#pragma unroll
    for (int i = 0; i < 8; ++i) { a[i] = lo[i]; a[i + 8] = hi[i]; }
#pragma unroll
    for (int j = 0; j < 4; ++j) {
      v16bf bf = load_b_bf16(Bt + (size_t)(j * 16) * kT + k0, kT, lane);
      acc[j] = wmma_bf16(a, bf, acc[j]);
    }
  }

  // spill per-wave partials into the dead logit strip and reduce
  float* part = s_logits;                  // reuse: partial[wave][row][d]
#pragma unroll
  for (int j = 0; j < 4; ++j)
#pragma unroll
    for (int r = 0; r < 8; ++r)
      part[wave * 1024 + (r + half * 8) * 64 + j * 16 + col] = acc[j][r];
  __syncthreads();

  for (int o = threadIdx.x; o < 1024; o += 256) {
    float sres = 0.0f;
#pragma unroll
    for (int w = 0; w < 8; ++w) sres += part[w * 1024 + o];
    const int row = o >> 6;
    const int d = o & 63;
    ypre[(size_t)(b * kT + i0 + row) * kC + h * 64 + d] = (bf16_t)sres;
  }
}

// ---- stage 3: y = y_pre @ W_proj + b_proj ---------------------------------
__global__ void __launch_bounds__(256) proj_gemm_kernel(
    const bf16_t* __restrict__ ypre, const bf16_t* __restrict__ WpT,
    const float* __restrict__ b_proj, float* __restrict__ y) {
  const int lane = threadIdx.x & 31;
  const int widx = blockIdx.x * 8 + (threadIdx.x >> 5);
  const int tn = widx & 15;            // 1024/64
  const int tm = widx >> 4;            // 8192/16
  const int m0 = tm * 16, n0 = tn * 64;
  const bf16_t* A = ypre + (size_t)m0 * kC;

  v8f acc[4];
#pragma unroll
  for (int j = 0; j < 4; ++j) acc[j] = vzero8();

  for (int k0 = 0; k0 < kC; k0 += 32) {
    __builtin_prefetch(A + k0 + 256, 0, 1);
    v16bf a = load_a_bf16(A + k0, kC, lane);
#pragma unroll
    for (int j = 0; j < 4; ++j) {
      v16bf b = load_b_bf16(WpT + (size_t)(n0 + j * 16) * kC + k0, kC, lane);
      acc[j] = wmma_bf16(a, b, acc[j]);
    }
  }

  const int col = lane & 15, half = lane >> 4;
#pragma unroll
  for (int j = 0; j < 4; ++j) {
    const int n = n0 + j * 16 + col;
    const float bias = b_proj[n];
#pragma unroll
    for (int r = 0; r < 8; ++r) {
      const int m = m0 + r + half * 8;
      y[(size_t)m * kC + n] = acc[j][r] + bias;
    }
  }
}

// ---------------------------------------------------------------------------

extern "C" void kernel_launch(void* const* d_in, const int* in_sizes, int n_in,
                              void* d_out, int out_size, void* d_ws, size_t ws_size,
                              hipStream_t stream) {
  (void)in_sizes; (void)n_in; (void)out_size; (void)ws_size;

  const float* x  = (const float*)d_in[0];
  const unsigned char* pad = (const unsigned char*)d_in[1];   // bool mask [B,1,T]
  const float* Wa = (const float*)d_in[2];
  const float* ba = (const float*)d_in[3];
  const float* Wp = (const float*)d_in[4];
  const float* bp = (const float*)d_in[5];

  float* y   = (float*)d_out;                 // [B,T,C]
  float* att = (float*)d_out + kYElems;       // [H,B,T,T]

  char* ws = (char*)d_ws;
  bf16_t* xb  = (bf16_t*)(ws + OFF_XB);
  bf16_t* WaT = (bf16_t*)(ws + OFF_WA);
  bf16_t* WpT = (bf16_t*)(ws + OFF_WP);
  bf16_t* qb  = (bf16_t*)(ws + OFF_Q);
  bf16_t* kb  = (bf16_t*)(ws + OFF_K);
  bf16_t* vT  = (bf16_t*)(ws + OFF_VT);
  bf16_t* yp  = (bf16_t*)(ws + OFF_YP);

  // prep: bf16 conversion + weight transposes
  cvt_f32_to_bf16_kernel<<<(kM * kC + 255) / 256, 256, 0, stream>>>(x, xb, kM * kC);
  transpose_f32_to_bf16_kernel<<<(kC * kN3 + 255) / 256, 256, 0, stream>>>(Wa, WaT, kC, kN3);
  transpose_f32_to_bf16_kernel<<<(kC * kC + 255) / 256, 256, 0, stream>>>(Wp, WpT, kC, kC);

  // stage 1: QKV (512 * 48 wave-tiles, 8 waves/block)
  qkv_gemm_kernel<<<(512 * 48) / 8, 256, 0, stream>>>(xb, WaT, ba, qb, kb, vT);

  // stage 2: fused attention core, one block per (bh, 16-row tile)
  fused_attn_kernel<<<64 * 128, 256, 0, stream>>>(qb, kb, vT, pad, att, yp);

  // stage 3: output projection (512 * 16 wave-tiles)
  proj_gemm_kernel<<<(512 * 16) / 8, 256, 0, stream>>>(yp, WpT, bp, y);
}
